// CEqGINO_69930657513594
// MI455X (gfx1250) — matrix operand order repaired and checked by reading, not compiled, not run on hardware
//
#include <hip/hip_runtime.h>
#include <hip/hip_bf16.h>
#include <math.h>

typedef __attribute__((ext_vector_type(16))) _Float16 v16h;
typedef __attribute__((ext_vector_type(8)))  float    v8f;
typedef __attribute__((ext_vector_type(2)))  _Float16 h2;

__device__ __forceinline__ float geluf(float x) {
  return 0.5f * x * (1.0f + erff(x * 0.70710678118654752440f));
}

// ---- WMMA operand loaders (CDNA5 16x16x32 f16 layouts, ISA 7.12.2) ----
// A: 16x32 f16 from row-major LDS. lane holds row M=lane&15; half=lane>>4;
// VGPR j holds K = (j>>2)*16 + half*8 + (j&3)*2 (+1).  (coalesces to 2x ds_load_b128)
__device__ __forceinline__ v16h load_A16(const _Float16* base, int lda, int lane) {
  int m = lane & 15, half = lane >> 4;
  const _Float16* row = base + m * lda;
  v16h a;
#pragma unroll
  for (int j = 0; j < 8; ++j) {
    int k = ((j >> 2) << 4) + (half << 3) + ((j & 3) << 1);
    h2 p = *reinterpret_cast<const h2*>(row + k);
    a[2 * j] = p.x;
    a[2 * j + 1] = p.y;
  }
  return a;
}

// Generic WMMA layer: 16 rows x (KSTEPS*32) -> 16 rows x 128.
// Weights pre-swizzled: tile (ks,n) is 512 contiguous f16, lane-major, so each
// B operand is one 32-byte vector load.
template <int KSTEPS>
__device__ __forceinline__ void wmma_layer(const _Float16* Abase, int lda,
                                           const _Float16* Wsw, const float* bias,
                                           _Float16* Out, int ldo, bool act, int lane) {
  int n16 = lane & 15, half = lane >> 4;
  v16h a[KSTEPS];
#pragma unroll
  for (int ks = 0; ks < KSTEPS; ++ks) a[ks] = load_A16(Abase + ks * 32, lda, lane);
#pragma unroll
  for (int n = 0; n < 8; ++n) {
    v8f c = {};
#pragma unroll
    for (int ks = 0; ks < KSTEPS; ++ks) {
      v16h b = reinterpret_cast<const v16h*>(Wsw + ((ks * 8 + n) << 9))[lane];
      c = __builtin_amdgcn_wmma_f32_16x16x32_f16(false, a[ks], false, b, (short)0, c,
                                                 false, false);
    }
    float bv = bias[n * 16 + n16];
#pragma unroll
    for (int r = 0; r < 8; ++r) {
      float v = c[r] + bv;
      if (act) v = geluf(v);
      Out[(r + half * 8) * ldo + n * 16 + n16] = (_Float16)v;
    }
  }
}

// ---- prep: convert f32 weight [K x 128] to f16, swizzled into B-operand tiles,
// zero-padded to Kpad rows. dst size = (Kpad/32)*8*512 f16.
__global__ void k_cvt_swz(const float* __restrict__ W, _Float16* __restrict__ dst,
                          int K, int Kpad) {
  int i = blockIdx.x * 256 + threadIdx.x;
  int total = (Kpad >> 5) * 8 * 512;
  if (i >= total) return;
  int j2 = i & 15;
  int l = (i >> 4) & 31;
  int tile = i >> 9;
  int tn = tile & 7, tk = tile >> 3;
  int k = tk * 32 + ((j2 >> 3) << 4) + ((l >> 4) << 3) + (j2 & 7);
  int n = tn * 16 + (l & 15);
  dst[i] = (k < K) ? (_Float16)W[k * 128 + n] : (_Float16)0.f;
}

__global__ void k_unorm(const float* __restrict__ u, float* __restrict__ un) {
  float x = u[0], y = u[1], z = u[2];
  float n = sqrtf(x * x + y * y + z * z) + 1e-8f;
  un[0] = x / n; un[1] = y / n; un[2] = z / n;
}

// ---- node encoder MLP (6->128->128->128, gelu,gelu,none) with WMMA ----
// block = 128 threads = 4 waves; each wave one 16-row tile; block covers 64 rows.
__global__ void k_node_enc(const float* __restrict__ fin,
                           const _Float16* __restrict__ W1, const float* __restrict__ B1,
                           const _Float16* __restrict__ W2, const float* __restrict__ B2,
                           const _Float16* __restrict__ W3, const float* __restrict__ B3,
                           _Float16* __restrict__ outF16, float* __restrict__ sumOut) {
  __shared__ _Float16 Ab[4][16][32];
  __shared__ _Float16 H0[4][16][128];
  __shared__ _Float16 H1[4][16][128];
  __shared__ float sbuf[128];
  int tid = threadIdx.x;
  int w = tid >> 5, lane = tid & 31;
  int r0 = blockIdx.x * 64 + w * 16;
  if (tid < 128) sbuf[tid] = 0.f;
  for (int i = lane; i < 512; i += 32) {
    int r = i >> 5, cc = i & 31;
    Ab[w][r][cc] = (cc < 6) ? (_Float16)fin[(size_t)(r0 + r) * 6 + cc] : (_Float16)0.f;
  }
  __syncthreads();
  wmma_layer<1>(&Ab[w][0][0], 32, W1, B1, &H0[w][0][0], 128, true, lane);
  __syncthreads();
  wmma_layer<4>(&H0[w][0][0], 128, W2, B2, &H1[w][0][0], 128, true, lane);
  __syncthreads();
  wmma_layer<4>(&H1[w][0][0], 128, W3, B3, &H0[w][0][0], 128, false, lane);
  __syncthreads();
  // write out + accumulate column sums for the mean
  int rb = blockIdx.x * 64;
  for (int i = tid; i < 64 * 128; i += 128) {
    int r = i >> 7, col = i & 127;
    float v = (float)H0[r >> 4][r & 15][col];
    outF16[(size_t)(rb + r) * 128 + col] = (_Float16)v;
    atomicAdd(&sbuf[col], v);
  }
  __syncthreads();
  if (tid < 128) atomicAdd(&sumOut[tid], sbuf[tid]);
}

// ---- lift KNN: per anchor, top-128 nearest of N points ----
// sorted top-set maintained in LDS; chunks that cannot contribute (all candidates
// >= current kth distance) skip the bitonic work entirely.
__global__ void k_lift_knn(const float* __restrict__ ac, const float* __restrict__ xc,
                           int N, float* __restrict__ d2o, int* __restrict__ idxo) {
  int m = blockIdx.x, t = threadIdx.x;  // 128 threads
  __shared__ float bd[256];
  __shared__ int bi[256];
  __shared__ int anyFlag;
  float ax = ac[m * 3 + 0], ay = ac[m * 3 + 1], az = ac[m * 3 + 2];
  bd[t] = 3.0e38f; bi[t] = 0;
  __syncthreads();
  int nch = N >> 7;
  for (int ch = 0; ch < nch; ++ch) {
    int pidx = (ch << 7) + t;
    float dx = xc[pidx * 3 + 0] - ax, dy = xc[pidx * 3 + 1] - ay, dz = xc[pidx * 3 + 2] - az;
    float cand = dx * dx + dy * dy + dz * dz;
    float thresh = bd[127];
    if (t == 0) anyFlag = 0;
    __syncthreads();
    if (cand < thresh) anyFlag = 1;
    __syncthreads();
    if (anyFlag) {
      bd[128 + t] = cand;
      bi[128 + t] = pidx;
      // sort new candidates ascending
      for (int ksz = 2; ksz <= 128; ksz <<= 1)
        for (int j = ksz >> 1; j > 0; j >>= 1) {
          __syncthreads();
          int ixj = t ^ j;
          if (ixj > t) {
            bool up = ((t & ksz) == 0);
            float va = bd[128 + t], vb = bd[128 + ixj];
            if ((va > vb) == up) {
              bd[128 + t] = vb; bd[128 + ixj] = va;
              int ti = bi[128 + t]; bi[128 + t] = bi[128 + ixj]; bi[128 + ixj] = ti;
            }
          }
        }
      __syncthreads();
      // keep 128 smallest of (best asc, new asc): C[i]=min(A[i],B[127-i]) is bitonic
      float nv = bd[255 - t];
      int ni = bi[255 - t];
      if (nv < bd[t]) { bd[t] = nv; bi[t] = ni; }
      // cleanup merge to ascending
      for (int j = 64; j > 0; j >>= 1) {
        __syncthreads();
        int ixj = t ^ j;
        if (ixj > t) {
          float va = bd[t], vb = bd[ixj];
          if (va > vb) {
            bd[t] = vb; bd[ixj] = va;
            int ti = bi[t]; bi[t] = bi[ixj]; bi[ixj] = ti;
          }
        }
      }
      __syncthreads();
    }
  }
  d2o[m * 128 + t] = bd[t];
  idxo[m * 128 + t] = bi[t];
}

// ---- decoder KNN: per query, full bitonic sort of 512 anchor distances ----
__global__ void k_dec_knn(const float* __restrict__ qc, const float* __restrict__ ac, int M,
                          float* __restrict__ d48, int* __restrict__ i48) {
  int q = blockIdx.x, t = threadIdx.x;  // 256 threads
  __shared__ float ds[512];
  __shared__ int is[512];
  float qx = qc[q * 3 + 0], qy = qc[q * 3 + 1], qz = qc[q * 3 + 2];
  for (int i = t; i < 512; i += 256) {
    if (i < M) {
      float dx = ac[i * 3 + 0] - qx, dy = ac[i * 3 + 1] - qy, dz = ac[i * 3 + 2] - qz;
      ds[i] = dx * dx + dy * dy + dz * dz;
    } else ds[i] = 3.0e38f;
    is[i] = i;
  }
  for (int ksz = 2; ksz <= 512; ksz <<= 1)
    for (int j = ksz >> 1; j > 0; j >>= 1) {
      __syncthreads();
      int idx = ((t & ~(j - 1)) << 1) | (t & (j - 1));
      int p = idx | j;
      bool up = ((idx & ksz) == 0);
      float va = ds[idx], vb = ds[p];
      if ((va > vb) == up) {
        ds[idx] = vb; ds[p] = va;
        int ti = is[idx]; is[idx] = is[p]; is[p] = ti;
      }
    }
  __syncthreads();
  if (t < 48) { d48[q * 48 + t] = ds[t]; i48[q * 48 + t] = is[t]; }
}

// ---- generic local-aggregation branch; NQ centers per block ----
// edge in = [feat(128), d2, c.u, n.u, (c-n).u, cond(16)] = 148, padded to 160.
// MLP 160->128->128->128 (gelu,gelu,none), gate 128->128 (gelu) + 128->1,
// softmax(logits - d2), weighted sum of eh.
template <int KN, int NQ>
__global__ void k_branch(const float* __restrict__ centers, const float* __restrict__ ncoords,
                         const _Float16* __restrict__ feats,
                         const int* __restrict__ kidx, const float* __restrict__ kd2, int kstride,
                         const float* __restrict__ u, const float* __restrict__ cond,
                         const _Float16* __restrict__ W1, const float* __restrict__ B1,
                         const _Float16* __restrict__ W2, const float* __restrict__ B2,
                         const _Float16* __restrict__ W3, const float* __restrict__ B3,
                         const _Float16* __restrict__ GW1, const float* __restrict__ GB1,
                         const float* __restrict__ gw2, const float* __restrict__ gb2,
                         float* __restrict__ aggOut) {
  constexpr int NW = KN / 16;
  constexpr int TPQ = NW * 32;
  __shared__ _Float16 ein[NQ][KN][160];
  __shared__ _Float16 hb[NQ][KN][128];
  __shared__ float dsm[NQ][KN];
  __shared__ float wsm[NQ][KN];
  int tid = threadIdx.x;
  int qloc = tid / TPQ;
  int lt = tid % TPQ;
  int c = blockIdx.x * NQ + qloc;
  if (tid == 0) {
    __builtin_prefetch(W1, 0, 0);
    __builtin_prefetch(W2, 0, 0);
    __builtin_prefetch(W3, 0, 0);
    __builtin_prefetch(GW1, 0, 0);
  }
  float ux = u[0], uy = u[1], uz = u[2];
  if (lt < KN) {
    float cx = centers[c * 3 + 0], cy = centers[c * 3 + 1], cz = centers[c * 3 + 2];
    float cdu = cx * ux + cy * uy + cz * uz;
    int nb = kidx[(size_t)c * kstride + lt];
    float d2 = kd2[(size_t)c * kstride + lt];
    dsm[qloc][lt] = d2;
    float nx = ncoords[nb * 3 + 0], ny = ncoords[nb * 3 + 1], nz = ncoords[nb * 3 + 2];
    float ndu = nx * ux + ny * uy + nz * uz;
    // vectorized 256B feature copy (16 x b128)
    const int4* f4 = reinterpret_cast<const int4*>(feats + (size_t)nb * 128);
    int4* e4 = reinterpret_cast<int4*>(&ein[qloc][lt][0]);
#pragma unroll
    for (int j = 0; j < 16; ++j) e4[j] = f4[j];
    ein[qloc][lt][128] = (_Float16)d2;
    ein[qloc][lt][129] = (_Float16)cdu;
    ein[qloc][lt][130] = (_Float16)ndu;
    ein[qloc][lt][131] = (_Float16)(cdu - ndu);
#pragma unroll
    for (int j = 0; j < 16; ++j) ein[qloc][lt][132 + j] = (_Float16)cond[j];
#pragma unroll
    for (int j = 148; j < 160; ++j) ein[qloc][lt][j] = (_Float16)0.f;
  }
  __syncthreads();
  int w = lt >> 5, lane = tid & 31;
  wmma_layer<5>(&ein[qloc][w * 16][0], 160, W1, B1, &hb[qloc][w * 16][0], 128, true, lane);
  __syncthreads();
  wmma_layer<4>(&hb[qloc][w * 16][0], 128, W2, B2, &ein[qloc][w * 16][0], 160, true, lane);
  __syncthreads();
  wmma_layer<4>(&ein[qloc][w * 16][0], 160, W3, B3, &hb[qloc][w * 16][0], 128, false, lane);  // eh
  __syncthreads();
  wmma_layer<4>(&hb[qloc][w * 16][0], 128, GW1, GB1, &ein[qloc][w * 16][0], 160, true, lane);  // gate hid
  __syncthreads();
  if (lt < KN) {
    float s = gb2[0];
    for (int j = 0; j < 128; ++j) s += (float)ein[qloc][lt][j] * gw2[j];
    wsm[qloc][lt] = s - dsm[qloc][lt];
  }
  __syncthreads();
  if (lt == 0) {
    float mx = wsm[qloc][0];
    for (int e = 1; e < KN; ++e) mx = fmaxf(mx, wsm[qloc][e]);
    float sum = 0.f;
    for (int e = 0; e < KN; ++e) { float ex = expf(wsm[qloc][e] - mx); wsm[qloc][e] = ex; sum += ex; }
    float inv = 1.f / sum;
    for (int e = 0; e < KN; ++e) wsm[qloc][e] *= inv;
  }
  __syncthreads();
  for (int hd = lt; hd < 128; hd += TPQ) {
    float s = 0.f;
    for (int e = 0; e < KN; ++e) s += wsm[qloc][e] * (float)hb[qloc][e][hd];
    aggOut[(size_t)c * 128 + hd] = s;
  }
}

// ---- small scalar MLPs ----
// global context: MLP([sum*scale, cond(16)]) 144->128->128->128
__global__ void k_gctx(const float* __restrict__ sum, float scale, const float* __restrict__ cond,
                       const float* __restrict__ W1, const float* __restrict__ B1,
                       const float* __restrict__ W2, const float* __restrict__ B2,
                       const float* __restrict__ W3, const float* __restrict__ B3,
                       float* __restrict__ out) {
  __shared__ float xin[144], h1m[128], h2m[128];
  int t = threadIdx.x;
  if (t < 128) xin[t] = sum[t] * scale;
  if (t < 16) xin[128 + t] = cond[t];
  __syncthreads();
  if (t < 128) { float s = B1[t]; for (int k = 0; k < 144; ++k) s += xin[k] * W1[k * 128 + t]; h1m[t] = geluf(s); }
  __syncthreads();
  if (t < 128) { float s = B2[t]; for (int k = 0; k < 128; ++k) s += h1m[k] * W2[k * 128 + t]; h2m[t] = geluf(s); }
  __syncthreads();
  if (t < 128) { float s = B3[t]; for (int k = 0; k < 128; ++k) s += h2m[k] * W3[k * 128 + t]; out[t] = s; }
}

// lift out: h = MLP([agg, gctx, cond]) 272->128->128->128; also sum for dec global
__global__ void k_lift_out(const float* __restrict__ agg, const float* __restrict__ gctx,
                           const float* __restrict__ cond,
                           const float* __restrict__ W1, const float* __restrict__ B1,
                           const float* __restrict__ W2, const float* __restrict__ B2,
                           const float* __restrict__ W3, const float* __restrict__ B3,
                           _Float16* __restrict__ hOutH, float* __restrict__ hSum) {
  __shared__ float xin[272], h1m[128], h2m[128];
  int m = blockIdx.x, t = threadIdx.x;
  if (t < 128) { xin[t] = agg[(size_t)m * 128 + t]; xin[128 + t] = gctx[t]; }
  if (t < 16) xin[256 + t] = cond[t];
  __syncthreads();
  if (t < 128) { float s = B1[t]; for (int k = 0; k < 272; ++k) s += xin[k] * W1[k * 128 + t]; h1m[t] = geluf(s); }
  __syncthreads();
  if (t < 128) { float s = B2[t]; for (int k = 0; k < 128; ++k) s += h1m[k] * W2[k * 128 + t]; h2m[t] = geluf(s); }
  __syncthreads();
  if (t < 128) {
    float s = B3[t];
    for (int k = 0; k < 128; ++k) s += h2m[k] * W3[k * 128 + t];
    hOutH[(size_t)m * 128 + t] = (_Float16)s;
    atomicAdd(&hSum[t], s);
  }
}

// final: out = MLP([agg_s, agg_l, g, cond, q.u]) 401->128->128->4
__global__ void k_dec_out(const float* __restrict__ aggs, const float* __restrict__ aggl,
                          const float* __restrict__ g, const float* __restrict__ cond,
                          const float* __restrict__ qc, const float* __restrict__ u,
                          const float* __restrict__ W1, const float* __restrict__ B1,
                          const float* __restrict__ W2, const float* __restrict__ B2,
                          const float* __restrict__ W3, const float* __restrict__ B3,
                          float* __restrict__ out) {
  __shared__ float xin[401], h1m[128], h2m[128];
  int q = blockIdx.x, t = threadIdx.x;
  if (t < 128) {
    xin[t] = aggs[(size_t)q * 128 + t];
    xin[128 + t] = aggl[(size_t)q * 128 + t];
    xin[256 + t] = g[t];
  }
  if (t < 16) xin[384 + t] = cond[t];
  if (t == 0) xin[400] = qc[q * 3] * u[0] + qc[q * 3 + 1] * u[1] + qc[q * 3 + 2] * u[2];
  __syncthreads();
  if (t < 128) { float s = B1[t]; for (int k = 0; k < 401; ++k) s += xin[k] * W1[k * 128 + t]; h1m[t] = geluf(s); }
  __syncthreads();
  if (t < 128) { float s = B2[t]; for (int k = 0; k < 128; ++k) s += h1m[k] * W2[k * 128 + t]; h2m[t] = geluf(s); }
  __syncthreads();
  if (t < 4) { float s = B3[t]; for (int k = 0; k < 128; ++k) s += h2m[k] * W3[k * 4 + t]; out[(size_t)q * 4 + t] = s; }
}

extern "C" void kernel_launch(void* const* d_in, const int* in_sizes, int n_in,
                              void* d_out, int out_size, void* d_ws, size_t ws_size,
                              hipStream_t stream) {
  (void)n_in; (void)out_size; (void)ws_size;
  const float* in_coords = (const float*)d_in[0];
  const float* in_x = (const float*)d_in[1];
  const float* a_coords = (const float*)d_in[2];
  const float* q_coords = (const float*)d_in[3];
  const float* flow = (const float*)d_in[4];
  const float* cond = (const float*)d_in[5];
  auto Wf = [&](int i) { return (const float*)d_in[i]; };

  int N = in_sizes[0] / 3;
  int M = in_sizes[2] / 3;
  int Q = in_sizes[3] / 3;

  unsigned char* p = (unsigned char*)d_ws;
  auto carve = [&](size_t bytes) -> void* {
    void* r = (void*)p;
    p += (bytes + 255) & ~(size_t)255;
    return r;
  };
  float* u_norm = (float*)carve(4 * sizeof(float));
  float* nodeSum = (float*)carve(128 * sizeof(float));
  float* hSum = (float*)carve(128 * sizeof(float));
  float* gctx = (float*)carve(128 * sizeof(float));
  float* gdec = (float*)carve(128 * sizeof(float));
  _Float16* nodeF16 = (_Float16*)carve((size_t)N * 128 * 2);
  int* liftIdx = (int*)carve((size_t)M * 128 * 4);
  float* liftD2 = (float*)carve((size_t)M * 128 * 4);
  float* aggM = (float*)carve((size_t)M * 128 * 4);
  _Float16* hF16 = (_Float16*)carve((size_t)M * 128 * 2);
  int* decIdx = (int*)carve((size_t)Q * 48 * 4);
  float* decD2 = (float*)carve((size_t)Q * 48 * 4);
  float* aggS = (float*)carve((size_t)Q * 128 * 4);
  float* aggL = (float*)carve((size_t)Q * 128 * 4);
  auto carveW = [&](int Kpad) { return (_Float16*)carve((size_t)Kpad * 128 * 2); };
  _Float16* ne_W1 = carveW(32);
  _Float16* ne_W2 = carveW(128);
  _Float16* ne_W3 = carveW(128);
  _Float16* le_W1 = carveW(160);
  _Float16* le_W2 = carveW(128);
  _Float16* le_W3 = carveW(128);
  _Float16* lg_W1 = carveW(128);
  _Float16* ds_W1 = carveW(160);
  _Float16* ds_W2 = carveW(128);
  _Float16* ds_W3 = carveW(128);
  _Float16* gs_W1 = carveW(128);
  _Float16* dl_W1 = carveW(160);
  _Float16* dl_W2 = carveW(128);
  _Float16* dl_W3 = carveW(128);
  _Float16* gl_W1 = carveW(128);

  hipMemsetAsync(nodeSum, 0, 128 * sizeof(float), stream);
  hipMemsetAsync(hSum, 0, 128 * sizeof(float), stream);
  k_unorm<<<1, 1, 0, stream>>>(flow, u_norm);

  auto cvt = [&](int i, _Float16* dst, int K, int Kpad) {
    int total = Kpad * 128;
    k_cvt_swz<<<(total + 255) / 256, 256, 0, stream>>>((const float*)d_in[i], dst, K, Kpad);
  };
  cvt(6, ne_W1, 6, 32);    cvt(8, ne_W2, 128, 128);  cvt(10, ne_W3, 128, 128);
  cvt(12, le_W1, 148, 160); cvt(14, le_W2, 128, 128); cvt(16, le_W3, 128, 128);
  cvt(18, lg_W1, 128, 128);
  cvt(34, ds_W1, 148, 160); cvt(36, ds_W2, 128, 128); cvt(38, ds_W3, 128, 128);
  cvt(40, gs_W1, 128, 128);
  cvt(44, dl_W1, 148, 160); cvt(46, dl_W2, 128, 128); cvt(48, dl_W3, 128, 128);
  cvt(50, gl_W1, 128, 128);

  // node encoder
  k_node_enc<<<N / 64, 128, 0, stream>>>(in_x, ne_W1, Wf(7), ne_W2, Wf(9), ne_W3, Wf(11),
                                         nodeF16, nodeSum);
  // lift KNN + edge aggregation
  k_lift_knn<<<M, 128, 0, stream>>>(a_coords, in_coords, N, liftD2, liftIdx);
  k_branch<128, 1><<<M, 256, 0, stream>>>(a_coords, in_coords, nodeF16, liftIdx, liftD2, 128,
                                          u_norm, cond, le_W1, Wf(13), le_W2, Wf(15), le_W3, Wf(17),
                                          lg_W1, Wf(19), Wf(20), Wf(21), aggM);
  // global ctx + anchor features
  k_gctx<<<1, 128, 0, stream>>>(nodeSum, 1.0f / (float)N, cond, Wf(22), Wf(23), Wf(24), Wf(25),
                                Wf(26), Wf(27), gctx);
  k_lift_out<<<M, 128, 0, stream>>>(aggM, gctx, cond, Wf(28), Wf(29), Wf(30), Wf(31), Wf(32),
                                    Wf(33), hF16, hSum);
  k_gctx<<<1, 128, 0, stream>>>(hSum, 1.0f / (float)M, cond, Wf(54), Wf(55), Wf(56), Wf(57),
                                Wf(58), Wf(59), gdec);
  // decoder KNN + branches (8 queries/block for k=16, 2 for k=48)
  k_dec_knn<<<Q, 256, 0, stream>>>(q_coords, a_coords, M, decD2, decIdx);
  k_branch<16, 8><<<Q / 8, 256, 0, stream>>>(q_coords, a_coords, hF16, decIdx, decD2, 48,
                                             u_norm, cond, ds_W1, Wf(35), ds_W2, Wf(37),
                                             ds_W3, Wf(39), gs_W1, Wf(41), Wf(42), Wf(43), aggS);
  k_branch<48, 2><<<Q / 2, 192, 0, stream>>>(q_coords, a_coords, hF16, decIdx, decD2, 48,
                                             u_norm, cond, dl_W1, Wf(45), dl_W2, Wf(47),
                                             dl_W3, Wf(49), gl_W1, Wf(51), Wf(52), Wf(53), aggL);
  // final out MLP
  k_dec_out<<<Q, 128, 0, stream>>>(aggS, aggL, gdec, cond, q_coords, u_norm,
                                   Wf(60), Wf(61), Wf(62), Wf(63), Wf(64), Wf(65),
                                   (float*)d_out);
}
